// ATL_Layer_19284403159353
// MI455X (gfx1250) — matrix operand order repaired and checked by reading, not compiled, not run on hardware
//
#include <hip/hip_runtime.h>

typedef __attribute__((ext_vector_type(16))) _Float16 v16h;
typedef __attribute__((ext_vector_type(8)))  float    v8f;

#define T_        2
#define WQ_       75
#define WS_       25
#define C_        640
#define HW_       100
#define WAY_      5
#define SHOT_     5
#define HID_      40
#define NIMG_     (WQ_ + WS_)        // 100 images per t
#define ROWS_     (WQ_ * HW_)        // 7500
#define ROWS_PAD_ 7504               // padded to /16
#define COLS_     (WS_ * HW_)        // 2500
#define COLS_PAD_ 2512               // padded to /16
#define KST_      (C_ / 32)          // 20 K-steps of 32

union Frag {
    v16h v;
    _Float16 h[16];
    uint4 q[2];
};

__device__ __forceinline__ float sigmoidf_(float x) { return 1.0f / (1.0f + __expf(-x)); }
__device__ __forceinline__ float leakyf_(float x)   { return x >= 0.0f ? x : 0.2f * x; }

// -------------------------------------------------------------------------
// Kernel 1: embedded feature GEMM per image: Y[p][o] = leaky(BN(W @ X))[o,p]
// A = W' (BN scale folded into rows, f32->f16), B = X (f32->f16), WMMA f32 acc.
// Padded B columns use a clamped address (no divergent per-element loads);
// their garbage outputs are simply never stored.
// -------------------------------------------------------------------------
__global__ __launch_bounds__(256)
void embed_gemm(const float* __restrict__ qf, const float* __restrict__ sf,
                const float* __restrict__ Wc, const float* __restrict__ gamma,
                const float* __restrict__ beta, const float* __restrict__ mean,
                const float* __restrict__ var, float* __restrict__ Yt)
{
    int blk = blockIdx.x;                 // t*NIMG_ + img
    int t = blk / NIMG_, img = blk % NIMG_;
    const float* X = (img < WQ_)
        ? qf + (size_t)(t * WQ_ + img) * C_ * HW_
        : sf + (size_t)(t * WS_ + (img - WQ_)) * C_ * HW_;
    float* Yrow = Yt + (size_t)blk * HW_ * C_;    // [p][o]

    int tid = threadIdx.x, wave = tid >> 5, lane = tid & 31;
    int g = lane >> 4, l16 = lane & 15;

    for (int tt = wave; tt < 40 * 7; tt += 8) {   // 40 o-tiles x 7 p-tiles
        int ot = tt / 7, pt = tt % 7;
        int obase = ot * 16, pbase = pt * 16;
        int oA = obase + l16;                     // A-fragment row for this lane
        float invA = gamma[oA] * rsqrtf(var[oA] + 1e-5f);
        int pB = pbase + l16;                     // B-fragment column
        bool pv = pB < HW_;
        int pBs = pv ? pB : (HW_ - 1);            // clamped: no divergent loads

        v8f acc;
        #pragma unroll
        for (int i = 0; i < 8; ++i) acc[i] = 0.0f;

        for (int kt = 0; kt < KST_; ++kt) {
            Frag a, b;
            // A (16x32 f16): lane group g holds K = g*8+{0..7} and 16+g*8+{0..7}
            int cA0 = kt * 32 + g * 8;
            const float* wrow = Wc + (size_t)oA * C_;
            #pragma unroll
            for (int i = 0; i < 8; ++i) {
                a.h[i]     = (_Float16)(wrow[cA0 + i]      * invA);
                a.h[i + 8] = (_Float16)(wrow[cA0 + 16 + i] * invA);
            }
            // B (32x16 f16): lane group g holds K = g*8+{0..7} and 16+g*8+{0..7}, lane = column
            int cB0 = kt * 32 + g * 8;
            const float* xcol = X + pBs;
            #pragma unroll
            for (int i = 0; i < 16; ++i)
                b.h[i] = (_Float16)xcol[(size_t)(cB0 + i + (i >= 8 ? 8 : 0)) * HW_];
            acc = __builtin_amdgcn_wmma_f32_16x16x32_f16(
                false, a.v, false, b.v, (short)0, acc, false, false);
        }
        if (pv) {
            #pragma unroll
            for (int v = 0; v < 8; ++v) {
                int oC = obase + v + 8 * g;       // C/D row = v + 8*(lane/16)
                float iv = gamma[oC] * rsqrtf(var[oC] + 1e-5f);
                float bs = beta[oC] - mean[oC] * iv;
                Yrow[(size_t)pB * C_ + oC] = leakyf_(acc[v] + bs);
            }
        }
    }
}

// -------------------------------------------------------------------------
// Kernel 2: per-(t,img,p) L2-normalize embedded + raw rows -> f16 matrices,
// and compute the AEA clamp value via the psi MLP (query rows only).
// One wave per location; 4 waves per block.
// -------------------------------------------------------------------------
__global__ __launch_bounds__(128)
void normalize_mlp(const float* __restrict__ qf, const float* __restrict__ sf,
                   const float* __restrict__ Yt,
                   const float* __restrict__ w1, const float* __restrict__ b1,
                   const float* __restrict__ w2, const float* __restrict__ b2,
                   _Float16* __restrict__ Aemb, _Float16* __restrict__ Araw,
                   _Float16* __restrict__ Bemb, _Float16* __restrict__ Braw,
                   float* __restrict__ cv)
{
    __shared__ float bufE[4][C_];
    __shared__ float bufR[4][C_];
    int tid = threadIdx.x, wi = tid >> 5, lane = tid & 31;
    int gw = blockIdx.x * 4 + wi;
    int t = gw / (NIMG_ * HW_);
    int rem = gw % (NIMG_ * HW_);
    int img = rem / HW_, p = rem % HW_;

    const float* yrow = Yt + ((size_t)(t * NIMG_ + img) * HW_ + p) * C_;
    const float* X = (img < WQ_)
        ? qf + (size_t)(t * WQ_ + img) * C_ * HW_
        : sf + (size_t)(t * WS_ + (img - WQ_)) * C_ * HW_;

    float ssE = 0.0f, ssR = 0.0f;
    for (int c = lane; c < C_; c += 32) {
        float ve = yrow[c];                bufE[wi][c] = ve; ssE += ve * ve;
        float vr = X[(size_t)c * HW_ + p]; bufR[wi][c] = vr; ssR += vr * vr;
    }
    #pragma unroll
    for (int off = 16; off > 0; off >>= 1) {
        ssE += __shfl_xor(ssE, off, 32);
        ssR += __shfl_xor(ssR, off, 32);
    }
    float invE = 1.0f / fmaxf(sqrtf(ssE), 1e-12f);
    float invR = 1.0f / fmaxf(sqrtf(ssR), 1e-12f);
    __syncthreads();   // all warps in a block share the same img (HW_ % 4 == 0)

    _Float16 *de, *dr; int r = 0;
    if (img < WQ_) {
        r = img * HW_ + p;
        de = Aemb + ((size_t)t * ROWS_PAD_ + r) * C_;
        dr = Araw + ((size_t)t * ROWS_PAD_ + r) * C_;
    } else {
        int k = (img - WQ_) * HW_ + p;
        de = Bemb + ((size_t)t * COLS_PAD_ + k) * C_;
        dr = Braw + ((size_t)t * COLS_PAD_ + k) * C_;
    }
    for (int j = lane; j < C_ / 2; j += 32) {
        union { _Float16 h[2]; unsigned u; } pe, pr;
        pe.h[0] = (_Float16)(bufE[wi][2 * j]     * invE);
        pe.h[1] = (_Float16)(bufE[wi][2 * j + 1] * invE);
        pr.h[0] = (_Float16)(bufR[wi][2 * j]     * invR);
        pr.h[1] = (_Float16)(bufR[wi][2 * j + 1] * invR);
        ((unsigned*)de)[j] = pe.u;
        ((unsigned*)dr)[j] = pr.u;
    }
    if (img < WQ_) {
        // psi MLP on the L2-normalized embedded row: 640 -> 40 -> 1.
        // Second hidden unit uses a clamped index so the hot loop has no
        // divergent branch; lanes >= 8 just discard their a2.
        float a1 = 0.0f, a2 = 0.0f;
        int j1 = lane;
        int j2 = 32 + (lane & 7);
        for (int c = 0; c < C_; ++c) {
            float v = bufE[wi][c] * invE;
            a1 += v * w1[c * HID_ + j1];
            a2 += v * w1[c * HID_ + j2];
        }
        float s = leakyf_(a1 + b1[j1]) * w2[j1];
        if (lane < 8) s += leakyf_(a2 + b1[j2]) * w2[j2];
        #pragma unroll
        for (int off = 16; off > 0; off >>= 1) s += __shfl_xor(s, off, 32);
        if (lane == 0)
            cv[(size_t)t * ROWS_PAD_ + r] = sigmoidf_(s + b2[0]) * 0.3f + 0.5f;
    }
}

// -------------------------------------------------------------------------
// Kernel 3: fused dual GEMM (f_x & match) + sigmoid gate + L1 norm + way sum.
// One block per 16 query rows; A rows staged in LDS, reused across 157 tiles.
// -------------------------------------------------------------------------
__global__ __launch_bounds__(128)
void score_gemm(const _Float16* __restrict__ Aemb, const _Float16* __restrict__ Araw,
                const _Float16* __restrict__ Bemb, const _Float16* __restrict__ Braw,
                const float* __restrict__ cv, float* __restrict__ out)
{
    __shared__ _Float16 sE[16][C_];
    __shared__ _Float16 sR[16][C_];
    __shared__ float cvs[16];
    __shared__ float aL1[16];
    __shared__ float aS[16][WAY_];

    int t = blockIdx.y, rblk = blockIdx.x;
    int tid = threadIdx.x, wave = tid >> 5, lane = tid & 31;
    int g = lane >> 4, l16 = lane & 15;

    // Stage 16 A-rows (embedded + raw), contiguous in global memory.
    const unsigned* gE = (const unsigned*)(Aemb + ((size_t)t * ROWS_PAD_ + rblk * 16) * C_);
    const unsigned* gR = (const unsigned*)(Araw + ((size_t)t * ROWS_PAD_ + rblk * 16) * C_);
    for (int i = tid; i < 16 * C_ / 2; i += 128) {
        ((unsigned*)sE)[i] = gE[i];
        ((unsigned*)sR)[i] = gR[i];
    }
    if (tid < 16) { cvs[tid] = cv[(size_t)t * ROWS_PAD_ + rblk * 16 + tid]; aL1[tid] = 0.0f; }
    if (tid < 16 * WAY_) ((float*)aS)[tid] = 0.0f;
    __syncthreads();

    for (int nt = wave; nt < COLS_PAD_ / 16; nt += 4) {
        int k0 = nt * 16;
        const _Float16* bErow = Bemb + ((size_t)t * COLS_PAD_ + k0 + l16) * C_;
        const _Float16* bRrow = Braw + ((size_t)t * COLS_PAD_ + k0 + l16) * C_;

        // Prefetch next tile's B rows (global_prefetch_b8).
        if (nt + 4 < COLS_PAD_ / 16) {
            __builtin_prefetch(Bemb + ((size_t)t * COLS_PAD_ + (nt + 4) * 16 + l16) * C_, 0, 1);
            __builtin_prefetch(Braw + ((size_t)t * COLS_PAD_ + (nt + 4) * 16 + l16) * C_, 0, 1);
        }

        v8f accF, accM;
        #pragma unroll
        for (int i = 0; i < 8; ++i) { accF[i] = 0.0f; accM[i] = 0.0f; }

        #pragma unroll 2
        for (int kt = 0; kt < KST_; ++kt) {
            int cA0 = kt * 32 + g * 8;    // A layout: K = g*8+{0..7}, 16+g*8+{0..7}
            int cB0 = kt * 32 + g * 8;    // B layout: K = g*8+{0..7}, 16+g*8+{0..7}
            Frag aE, aR, bE, bR;
            aE.q[0] = *(const uint4*)&sE[l16][cA0];
            aE.q[1] = *(const uint4*)&sE[l16][cA0 + 16];
            aR.q[0] = *(const uint4*)&sR[l16][cA0];
            aR.q[1] = *(const uint4*)&sR[l16][cA0 + 16];
            bE.q[0] = *(const uint4*)(bErow + cB0);
            bE.q[1] = *(const uint4*)(bErow + cB0 + 16);
            bR.q[0] = *(const uint4*)(bRrow + cB0);
            bR.q[1] = *(const uint4*)(bRrow + cB0 + 16);
            accF = __builtin_amdgcn_wmma_f32_16x16x32_f16(
                false, aE.v, false, bE.v, (short)0, accF, false, false);
            accM = __builtin_amdgcn_wmma_f32_16x16x32_f16(
                false, aR.v, false, bR.v, (short)0, accM, false, false);
        }

        // Epilogue: gate, mask padding, reduce over the 16 columns of the tile.
        int kcol = k0 + l16;
        bool valid = kcol < COLS_;
        int wfirst = (k0 < COLS_ - 1 ? k0 : COLS_ - 1) / (SHOT_ * HW_);
        int klast  = (k0 + 15 < COLS_ - 1 ? k0 + 15 : COLS_ - 1);
        int wlast  = klast / (SHOT_ * HW_);
        int wthis  = valid ? kcol / (SHOT_ * HW_) : -1;

        #pragma unroll
        for (int v = 0; v < 8; ++v) {
            int m = v + 8 * g;            // C/D row held by this lane/vgpr
            float clampv = 0.0f, gate = 0.0f;
            if (valid) {
                clampv = sigmoidf_(50.0f * (accF[v] - cvs[m]));
                gate   = clampv * accM[v];
            }
            float g0 = (wthis == wfirst) ? gate : 0.0f;
            float g1 = (wlast != wfirst && wthis == wlast) ? gate : 0.0f;
            #pragma unroll
            for (int off = 8; off > 0; off >>= 1) {
                clampv += __shfl_xor(clampv, off, 16);
                g0     += __shfl_xor(g0, off, 16);
                g1     += __shfl_xor(g1, off, 16);
            }
            if (l16 == 0) {
                atomicAdd(&aL1[m], clampv);
                atomicAdd(&aS[m][wfirst], g0);
                if (wlast != wfirst) atomicAdd(&aS[m][wlast], g1);
            }
        }
    }
    __syncthreads();

    if (tid < 16) {
        int r = rblk * 16 + tid;
        if (r < ROWS_) {
            int q = r / HW_;
            float scale = (30.0f / 500.0f) / fmaxf(aL1[tid], 1e-12f);
            #pragma unroll
            for (int w = 0; w < WAY_; ++w)
                atomicAdd(&out[(t * WQ_ + q) * WAY_ + w], aS[tid][w] * scale);
        }
    }
}

// -------------------------------------------------------------------------
extern "C" void kernel_launch(void* const* d_in, const int* in_sizes, int n_in,
                              void* d_out, int out_size, void* d_ws, size_t ws_size,
                              hipStream_t stream)
{
    (void)in_sizes; (void)n_in; (void)ws_size;
    const float* qf    = (const float*)d_in[0];
    const float* sf    = (const float*)d_in[1];
    const float* Wc    = (const float*)d_in[2];
    const float* gamma = (const float*)d_in[3];
    const float* beta  = (const float*)d_in[4];
    const float* mean  = (const float*)d_in[5];
    const float* var   = (const float*)d_in[6];
    const float* w1    = (const float*)d_in[7];
    const float* b1    = (const float*)d_in[8];
    const float* w2    = (const float*)d_in[9];
    const float* b2    = (const float*)d_in[10];
    float* out = (float*)d_out;

    char* ws = (char*)d_ws;
    size_t off = 0;
    auto salloc = [&](size_t bytes) -> char* {
        char* p = ws + off;
        off += (bytes + 255) & ~(size_t)255;
        return p;
    };
    float*    Yt   = (float*)   salloc((size_t)T_ * NIMG_ * HW_ * C_ * sizeof(float));
    _Float16* Aemb = (_Float16*)salloc((size_t)T_ * ROWS_PAD_ * C_ * sizeof(_Float16));
    _Float16* Araw = (_Float16*)salloc((size_t)T_ * ROWS_PAD_ * C_ * sizeof(_Float16));
    _Float16* Bemb = (_Float16*)salloc((size_t)T_ * COLS_PAD_ * C_ * sizeof(_Float16));
    _Float16* Braw = (_Float16*)salloc((size_t)T_ * COLS_PAD_ * C_ * sizeof(_Float16));
    float*    cv   = (float*)   salloc((size_t)T_ * ROWS_PAD_ * sizeof(float));

    hipMemsetAsync(d_out, 0, (size_t)out_size * sizeof(float), stream);

    embed_gemm<<<T_ * NIMG_, 256, 0, stream>>>(qf, sf, Wc, gamma, beta, mean, var, Yt);
    normalize_mlp<<<(T_ * NIMG_ * HW_) / 4, 128, 0, stream>>>(
        qf, sf, Yt, w1, b1, w2, b2, Aemb, Araw, Bemb, Braw, cv);
    score_gemm<<<dim3(ROWS_PAD_ / 16, T_), 128, 0, stream>>>(
        Aemb, Araw, Bemb, Braw, cv, out);
}
